// BreakthroughSNN_60000693125491
// MI455X (gfx1250) — matrix operand order repaired and checked by reading, not compiled, not run on hardware
//
#include <hip/hip_runtime.h>
#include <hip/hip_bf16.h>
#include <hip/hip_fp16.h>

// ---------------------------------------------------------------------------
// BreakthroughSNN forward for MI455X (gfx1250, wave32, WMMA).
// All matmuls use V_WMMA_F32_16X16X32_F16 (f16 x f16 -> f32 accum).
// Spikes are exactly {0,1} (exact in f16); weights are either pre-packed once
// per launch into fragment-native f16 tiles (if workspace allows) or
// converted f32->f16 in-register at load time.
// ---------------------------------------------------------------------------

typedef __attribute__((ext_vector_type(16))) _Float16 v16h;
typedef __attribute__((ext_vector_type(8)))  float    v8f;

#define BS_     1024          // B*S tokens
#define T_      8
#define DMODEL  256
#define DSTATE  128
#define NHEAD   4
#define DHEAD   32
#define NLAYER  2
#define VOCAB   32000
#define DECAY_F 0.6065306597126334f   // exp(-1/2)
#define ADAPT_F 0.1f
#define TARGET_F 0.02f

// ---------------- fragment loaders (wave32, 16x16x32 f16 WMMA) -------------

// A fragment: row-major act[M][lda] f16, tile at (row0, k0).
// lane 0-15 -> M=lane, lanes 16-31 -> M=lane-16 with K offset +16 (contiguous k = 16*hi + j).
__device__ __forceinline__ v16h frag_a_f16(const _Float16* A, int lda,
                                           int row0, int k0, int lane) {
  const int m = lane & 15, hi = (lane >> 4) & 1;
  const _Float16* p = A + (size_t)(row0 + m) * lda + k0;
  v16h f;
#pragma unroll
  for (int j = 0; j < 8; ++j) f[j] = p[hi * 16 + j];
#pragma unroll
  for (int j = 0; j < 8; ++j) f[8 + j] = p[hi * 16 + 8 + j];
  return f;
}

// B fragment from Bt = B^T stored row-major [n][ldb] in f16 (B[k][n]=Bt[n][k]).
__device__ __forceinline__ v16h frag_b_f16(const _Float16* Bt, int ldb,
                                           int n0, int k0, int lane) {
  const int n = lane & 15, hi = (lane >> 4) & 1;
  const _Float16* p = Bt + (size_t)(n0 + n) * ldb + k0 + hi * 16;
  v16h f;
#pragma unroll
  for (int j = 0; j < 16; ++j) f[j] = p[j];
  return f;
}

// Same, but source weights are f32; convert in-register to f16.
__device__ __forceinline__ v16h frag_b_f32cvt(const float* Wt, int ldw,
                                              int n0, int k0, int lane) {
  const int n = lane & 15, hi = (lane >> 4) & 1;
  const float* p = Wt + (size_t)(n0 + n) * ldw + k0 + hi * 16;
  v16h f;
#pragma unroll
  for (int j = 0; j < 16; ++j) f[j] = (_Float16)p[j];
  return f;
}

// B fragment from B stored row-major [k][stride] (e.g. V matrix): B[k][n].
__device__ __forceinline__ v16h frag_b_strided_f16(const _Float16* base, int stride,
                                                   int k0, int lane) {
  const int n = lane & 15, hi = (lane >> 4) & 1;
  const _Float16* p = base + (size_t)(k0 + hi * 16) * stride + n;
  v16h f;
#pragma unroll
  for (int j = 0; j < 16; ++j) f[j] = p[(size_t)j * stride];
  return f;
}

// ---------------- weight pre-pack: f32 [N][K] -> fragment-native f16 ---------
// One wave per 16x32 B-tile; tile stored as 512 contiguous halves so a wave's
// fragment load is one coalesced 32B load per lane (1KB contiguous per wave).
__global__ void pack_w_kernel(const float* __restrict__ W,
                              _Float16* __restrict__ P, int N, int K) {
  const int lane = threadIdx.x;
  const int tiles_k = K >> 5;
  const int tile = blockIdx.x;
  const int nt = tile / tiles_k, kt = tile % tiles_k;
  const int n = nt * 16 + (lane & 15);
  const int hi = (lane >> 4) & 1;
  const float* p = W + (size_t)n * K + kt * 32 + hi * 16;
  _Float16* q = P + (size_t)tile * 512 + lane * 16;
#pragma unroll
  for (int j = 0; j < 16; ++j) q[j] = (_Float16)p[j];
}

// ---------------- generic WMMA GEMM: out = act(f16) @ W^T [+bias][+addend] ---
// One wave per block; each wave computes a 16x64 output tile (A frag reused 4x).
template <bool OUT_F16, bool HAS_BIAS, bool HAS_ADDEND, bool PACKED_B>
__global__ void gemm_wmma_kernel(const _Float16* __restrict__ act,
                                 const float* __restrict__ W,      // f32 [N][K]
                                 const _Float16* __restrict__ Wp,  // packed f16
                                 const float* __restrict__ bias,
                                 const float* __restrict__ addend,
                                 float* __restrict__ outf,
                                 _Float16* __restrict__ outh,
                                 int M, int N, int K) {
  const int lane = threadIdx.x;
  const int row0 = blockIdx.y * 16;
  const int nb   = blockIdx.x * 64;
  const int tiles_k = K >> 5;
  v8f acc[4] = {};
  for (int k0 = 0; k0 < K; k0 += 32) {
    // prefetch next A chunk for this lane's row (global_prefetch_b8)
    if (k0 + 32 < K)
      __builtin_prefetch(act + (size_t)(row0 + (lane & 15)) * K + k0 + 32, 0, 1);
    v16h a = frag_a_f16(act, K, row0, k0, lane);
#pragma unroll
    for (int t = 0; t < 4; ++t) {
      v16h b;
      if (PACKED_B) {
        const size_t tile = (size_t)((nb >> 4) + t) * tiles_k + (k0 >> 5);
        b = *(const v16h*)(Wp + tile * 512 + lane * 16);
      } else {
        b = frag_b_f32cvt(W, K, nb + t * 16, k0, lane);
      }
      acc[t] = __builtin_amdgcn_wmma_f32_16x16x32_f16(
          false, a, false, b, (short)0, acc[t], false, false);
    }
  }
  const int nl = lane & 15, hi = (lane >> 4) & 1;
#pragma unroll
  for (int t = 0; t < 4; ++t) {
    const int n = nb + t * 16 + nl;
    const float bs = HAS_BIAS ? bias[n] : 0.0f;
#pragma unroll
    for (int r = 0; r < 8; ++r) {
      const int m = row0 + hi * 8 + r;
      float v = acc[t][r] + bs;
      if (HAS_ADDEND) v += addend[(size_t)m * N + n];
      if (OUT_F16) outh[(size_t)m * N + n] = (_Float16)v;
      else         outf[(size_t)m * N + n] = v;
    }
  }
}

template <bool OUT_F16, bool HAS_BIAS, bool HAS_ADDEND>
static inline void launch_gemm(bool packed,
                               const _Float16* act, const float* W,
                               const _Float16* Wp, const float* bias,
                               const float* addend, float* outf, _Float16* outh,
                               int M, int N, int K, hipStream_t stream) {
  dim3 grid(N / 64, M / 16);
  if (packed)
    gemm_wmma_kernel<OUT_F16, HAS_BIAS, HAS_ADDEND, true>
        <<<grid, 32, 0, stream>>>(act, W, Wp, bias, addend, outf, outh, M, N, K);
  else
    gemm_wmma_kernel<OUT_F16, HAS_BIAS, HAS_ADDEND, false>
        <<<grid, 32, 0, stream>>>(act, W, Wp, bias, addend, outf, outh, M, N, K);
}

// ---------------- attention: per (head, 16-row Q tile) wave ------------------
// scores = softmax(Q K^T / sqrt(32)) over all 1024 keys; att = scores @ V.
// Full score row tile kept in 32KB LDS (f16); QK^T and PV via WMMA.
__global__ void attn_kernel(const _Float16* __restrict__ q,
                            const _Float16* __restrict__ kv,   // [1024][256]: K | V
                            _Float16* __restrict__ att) {      // [1024][128]
  __shared__ _Float16 sP[16 * BS_];
  const int lane = threadIdx.x;
  const int row0 = blockIdx.x * 16;
  const int head = blockIdx.y;
  const int nl = lane & 15, hi = (lane >> 4) & 1;
  const float scale = 0.17677669529663687f;   // 1/sqrt(32)

  // Q fragment (K-dim = d_head = 32 = exactly one WMMA)
  v16h qf = frag_a_f16(q + head * DHEAD, DSTATE, row0, 0, lane);

  // scores for all 1024 keys, 16 at a time
  for (int kt = 0; kt < BS_ / 16; ++kt) {
    v16h kf = frag_b_f16(kv + head * DHEAD, DMODEL, kt * 16, 0, lane);
    v8f s = {};
    s = __builtin_amdgcn_wmma_f32_16x16x32_f16(
        false, qf, false, kf, (short)0, s, false, false);
#pragma unroll
    for (int r = 0; r < 8; ++r)
      sP[(size_t)(hi * 8 + r) * BS_ + kt * 16 + nl] = (_Float16)(s[r] * scale);
  }
  __syncthreads();

  // row-wise softmax (16 rows x 1024 keys), f32 math, f16 storage
  for (int r = 0; r < 16; ++r) {
    float mx = -3.0e38f;
    for (int c = lane; c < BS_; c += 32) mx = fmaxf(mx, (float)sP[r * BS_ + c]);
#pragma unroll
    for (int o = 16; o >= 1; o >>= 1) mx = fmaxf(mx, __shfl_xor(mx, o, 32));
    float sum = 0.0f;
    for (int c = lane; c < BS_; c += 32) {
      float e = __expf((float)sP[r * BS_ + c] - mx);
      sum += e;
      sP[r * BS_ + c] = (_Float16)e;
    }
#pragma unroll
    for (int o = 16; o >= 1; o >>= 1) sum += __shfl_xor(sum, o, 32);
    const float inv = 1.0f / sum;
    for (int c = lane; c < BS_; c += 32)
      sP[r * BS_ + c] = (_Float16)((float)sP[r * BS_ + c] * inv);
  }
  __syncthreads();

  // att tile = P[16 x 1024] @ V[1024 x 32] via WMMA, K chunks of 32 keys
  v8f o0 = {}, o1 = {};
  const _Float16* vbase = kv + DSTATE + head * DHEAD;
  for (int kc = 0; kc < BS_ / 32; ++kc) {
    v16h pf = frag_a_f16(sP, BS_, 0, kc * 32, lane);
    v16h v0 = frag_b_strided_f16(vbase + 0,  DMODEL, kc * 32, lane);
    v16h v1 = frag_b_strided_f16(vbase + 16, DMODEL, kc * 32, lane);
    o0 = __builtin_amdgcn_wmma_f32_16x16x32_f16(
        false, pf, false, v0, (short)0, o0, false, false);
    o1 = __builtin_amdgcn_wmma_f32_16x16x32_f16(
        false, pf, false, v1, (short)0, o1, false, false);
  }
#pragma unroll
  for (int r = 0; r < 8; ++r) {
    const int m = row0 + hi * 8 + r;
    att[(size_t)m * DSTATE + head * DHEAD + nl]      = (_Float16)o0[r];
    att[(size_t)m * DSTATE + head * DHEAD + 16 + nl] = (_Float16)o1[r];
  }
}

// ---------------- LIF pointwise + spike-count reduction ----------------------
__global__ void lif_kernel(const float* __restrict__ pot,
                           float* __restrict__ vmem,
                           const float* __restrict__ thr,
                           _Float16* __restrict__ spikes,
                           float* __restrict__ accum,
                           unsigned int* __restrict__ count,
                           int n, int do_acc) {
  const int i = blockIdx.x * blockDim.x + threadIdx.x;
  float s = 0.0f;
  if (i < n) {
    const float vp = vmem[i] * DECAY_F + pot[i];
    const float th = *thr;
    s = (vp - th >= 0.0f) ? 1.0f : 0.0f;
    vmem[i]   = vp * (1.0f - s);
    spikes[i] = (_Float16)s;
    if (do_acc) accum[i] += s;
  }
  unsigned long long m = __ballot(s != 0.0f);
  if ((threadIdx.x & 31) == 0)
    atomicAdd(count, (unsigned int)__popcll(m));
}

__global__ void thr_update_kernel(float* thr, unsigned int* count, float inv_n) {
  const float c = (float)(*count);
  *count = 0u;
  *thr = fmaxf(*thr + ADAPT_F * (c * inv_n - TARGET_F), 0.5f);
}

__global__ void init_scalars_kernel(float* ts, float* to,
                                    unsigned int* c1, unsigned int* c2) {
  *ts = 1.0f; *to = 1.0f; *c1 = 0u; *c2 = 0u;
}

__global__ void embed_kernel(const int* __restrict__ ids,
                             const float* __restrict__ emb,
                             _Float16* __restrict__ tok) {
  const int b = blockIdx.x, d = threadIdx.x;
  tok[(size_t)b * DMODEL + d] = (_Float16)emb[(size_t)ids[b] * DMODEL + d];
}

__global__ void mean_kernel(const float* __restrict__ acc,
                            _Float16* __restrict__ ti, int n) {
  const int i = blockIdx.x * blockDim.x + threadIdx.x;
  if (i < n) ti[i] = (_Float16)(acc[i] * (1.0f / (float)T_));
}

// ---------------------------------------------------------------------------
extern "C" void kernel_launch(void* const* d_in, const int* in_sizes, int n_in,
                              void* d_out, int out_size, void* d_ws, size_t ws_size,
                              hipStream_t stream) {
  (void)in_sizes; (void)n_in; (void)out_size;
  const int*   ids  = (const int*)d_in[0];
  const float* emb  = (const float*)d_in[1];
  const float* A    = (const float*)d_in[2];
  const float* C    = (const float*)d_in[3];
  const float* Wq   = (const float*)d_in[4];
  const float* bq   = (const float*)d_in[5];
  const float* Wkv  = (const float*)d_in[6];
  const float* bkv  = (const float*)d_in[7];
  const float* Wo   = (const float*)d_in[8];
  const float* bo   = (const float*)d_in[9];
  const float* Wout = (const float*)d_in[10];
  const float* bout = (const float*)d_in[11];
  float* logits = (float*)d_out;

  // workspace layout, 256B-aligned slices (~11 MB core + ~17 MB packed weights)
  char* w = (char*)d_ws;
  size_t off = 0;
  auto take = [&](size_t bytes) -> char* {
    char* p = w + off;
    off += (bytes + 255) & ~(size_t)255;
    return p;
  };
  _Float16* tok  = (_Float16*)take((size_t)BS_ * DMODEL * 2);
  _Float16* xbuf = (_Float16*)take((size_t)T_ * BS_ * DMODEL * 2);
  _Float16* hbuf = (_Float16*)take((size_t)BS_ * DSTATE * 2);
  float*    sv   = (float*)take((size_t)BS_ * DSTATE * 4);
  float*    ov   = (float*)take((size_t)BS_ * DMODEL * 4);
  float*    st   = (float*)take((size_t)BS_ * DSTATE * 4);
  _Float16* qb   = (_Float16*)take((size_t)BS_ * DSTATE * 2);
  _Float16* kvb  = (_Float16*)take((size_t)BS_ * DMODEL * 2);
  _Float16* attb = (_Float16*)take((size_t)BS_ * DSTATE * 2);
  float*    upd  = (float*)take((size_t)BS_ * DSTATE * 4);
  float*    opot = (float*)take((size_t)BS_ * DMODEL * 4);
  float*    accm = (float*)take((size_t)BS_ * DMODEL * 4);
  _Float16* tih  = (_Float16*)take((size_t)BS_ * DMODEL * 2);
  float*    ts   = (float*)take(4);
  float*    to   = (float*)take(4);
  unsigned* c1   = (unsigned*)take(4);
  unsigned* c2   = (unsigned*)take(4);

  // packed-weight region (used only if ws_size is large enough; ws_size is
  // fixed per harness, so the chosen path is identical on every call)
  _Float16* pA[NLAYER], *pWq[NLAYER], *pWo[NLAYER], *pWkv[NLAYER], *pC[NLAYER];
  for (int l = 0; l < NLAYER; ++l) {
    pA[l]   = (_Float16*)take((size_t)DSTATE * DSTATE * 2);
    pWq[l]  = (_Float16*)take((size_t)DSTATE * DSTATE * 2);
    pWo[l]  = (_Float16*)take((size_t)DSTATE * DSTATE * 2);
    pWkv[l] = (_Float16*)take((size_t)2 * DSTATE * DMODEL * 2);
    pC[l]   = (_Float16*)take((size_t)DMODEL * DSTATE * 2);
  }
  _Float16* pWout = (_Float16*)take((size_t)VOCAB * DMODEL * 2);
  const bool packed = (ws_size >= off);

  embed_kernel<<<BS_, DMODEL, 0, stream>>>(ids, emb, tok);
  hipMemsetAsync(accm, 0, (size_t)BS_ * DMODEL * 4, stream);

  if (packed) {
    for (int l = 0; l < NLAYER; ++l) {
      const size_t dd = (size_t)DSTATE * DSTATE;
      pack_w_kernel<<<(DSTATE/16)*(DSTATE/32), 32, 0, stream>>>(A  + l*dd, pA[l],  DSTATE, DSTATE);
      pack_w_kernel<<<(DSTATE/16)*(DSTATE/32), 32, 0, stream>>>(Wq + l*dd, pWq[l], DSTATE, DSTATE);
      pack_w_kernel<<<(DSTATE/16)*(DSTATE/32), 32, 0, stream>>>(Wo + l*dd, pWo[l], DSTATE, DSTATE);
      pack_w_kernel<<<(2*DSTATE/16)*(DMODEL/32), 32, 0, stream>>>(
          Wkv + (size_t)l*2*DSTATE*DMODEL, pWkv[l], 2*DSTATE, DMODEL);
      pack_w_kernel<<<(DMODEL/16)*(DSTATE/32), 32, 0, stream>>>(
          C + (size_t)l*DMODEL*DSTATE, pC[l], DMODEL, DSTATE);
    }
    pack_w_kernel<<<(VOCAB/16)*(DMODEL/32), 32, 0, stream>>>(Wout, pWout, VOCAB, DMODEL);
  }

  for (int l = 0; l < NLAYER; ++l) {
    hipMemsetAsync(hbuf, 0, (size_t)BS_ * DSTATE * 2, stream);
    hipMemsetAsync(sv,   0, (size_t)BS_ * DSTATE * 4, stream);
    hipMemsetAsync(ov,   0, (size_t)BS_ * DMODEL * 4, stream);
    init_scalars_kernel<<<1, 1, 0, stream>>>(ts, to, c1, c2);

    const float* Al   = A   + (size_t)l * DSTATE * DSTATE;
    const float* Cl   = C   + (size_t)l * DMODEL * DSTATE;
    const float* Wql  = Wq  + (size_t)l * DSTATE * DSTATE;
    const float* bql  = bq  + (size_t)l * DSTATE;
    const float* Wkvl = Wkv + (size_t)l * 2 * DSTATE * DMODEL;
    const float* bkvl = bkv + (size_t)l * 2 * DSTATE;
    const float* Wol  = Wo  + (size_t)l * DSTATE * DSTATE;
    const float* bol  = bo  + (size_t)l * DSTATE;

    for (int t = 0; t < T_; ++t) {
      const _Float16* xt = (l == 0) ? tok : (xbuf + (size_t)t * BS_ * DMODEL);

      // st = h @ A^T
      launch_gemm<false, false, false>(packed, hbuf, Al, pA[l], nullptr, nullptr,
                                       st, nullptr, BS_, DSTATE, DSTATE, stream);
      // q = h @ Wq^T + bq  (f16 out)
      launch_gemm<true, true, false>(packed, hbuf, Wql, pWq[l], bql, nullptr,
                                     nullptr, qb, BS_, DSTATE, DSTATE, stream);
      // kv = x_t @ Wkv^T + bkv  (f16 out, [K|V])
      launch_gemm<true, true, false>(packed, xt, Wkvl, pWkv[l], bkvl, nullptr,
                                     nullptr, kvb, BS_, 2 * DSTATE, DMODEL, stream);
      // attention across all 1024 positions, per head
      attn_kernel<<<dim3(BS_ / 16, NHEAD), 32, 0, stream>>>(qb, kvb, attb);
      // upd = st + att @ Wo^T + bo
      launch_gemm<false, true, true>(packed, attb, Wol, pWo[l], bol, st,
                                     upd, nullptr, BS_, DSTATE, DSTATE, stream);
      // LIF 1: state spikes -> h (f16), update sv, scalar threshold ts
      lif_kernel<<<(BS_ * DSTATE + 255) / 256, 256, 0, stream>>>(
          upd, sv, ts, hbuf, nullptr, c1, BS_ * DSTATE, 0);
      thr_update_kernel<<<1, 1, 0, stream>>>(ts, c1, 1.0f / (BS_ * DSTATE));
      // out_pot = h2 @ C^T
      launch_gemm<false, false, false>(packed, hbuf, Cl, pC[l], nullptr, nullptr,
                                       opot, nullptr, BS_, DMODEL, DSTATE, stream);
      // LIF 2: output spikes -> xbuf[t]; accumulate time-sum on last layer
      lif_kernel<<<(BS_ * DMODEL + 255) / 256, 256, 0, stream>>>(
          opot, ov, to, xbuf + (size_t)t * BS_ * DMODEL, accm, c2,
          BS_ * DMODEL, (l == NLAYER - 1) ? 1 : 0);
      thr_update_kernel<<<1, 1, 0, stream>>>(to, c2, 1.0f / (BS_ * DMODEL));
    }
  }

  // time_integrated = acc/T (exact in f16), then logits = ti @ Wout^T + bout
  mean_kernel<<<(BS_ * DMODEL + 255) / 256, 256, 0, stream>>>(accm, tih, BS_ * DMODEL);
  launch_gemm<false, true, false>(packed, tih, Wout, pWout, bout, nullptr,
                                  logits, nullptr, BS_, VOCAB, DMODEL, stream);
}
